// MultiHeadSelfAttention_87101936763570
// MI455X (gfx1250) — matrix operand order repaired and checked by reading, not compiled, not run on hardware
//
#include <hip/hip_runtime.h>
#include <hip/hip_bf16.h>
#include <stdint.h>

// ---------------------------------------------------------------------------
// MHA forward for MI455X (gfx1250): bf16 WMMA GEMMs + fused flash-attention,
// with async global->LDS double-buffered tile pipelines.
// ---------------------------------------------------------------------------

typedef unsigned short bfu;                                   // raw bf16 bits
typedef __attribute__((ext_vector_type(16))) __bf16 v16bf;    // WMMA A/B frag
typedef __attribute__((ext_vector_type(8)))  float  v8f;      // WMMA C/D frag
typedef int v4i __attribute__((vector_size(16)));             // 16B async unit

union FragB { v16bf v; uint4 q[2]; };   // 32 bytes: 16 bf16 elements

#define D_MODEL   1024
#define NUM_HEADS 16
#define BATCH     4
#define SEQ       2048
#define DK        64

__device__ __forceinline__ v8f v8f_zero() {
  v8f z;
#pragma unroll
  for (int i = 0; i < 8; i++) z[i] = 0.0f;
  return z;
}

__device__ __forceinline__ v8f wmma_bf16(v16bf a, v16bf b, v8f c) {
  // v_wmma_f32_16x16x32_bf16 : D = A(16x32) * B(32x16) + C(16x16 f32)
  return __builtin_amdgcn_wmma_f32_16x16x32_bf16(
      /*neg_a=*/false, a, /*neg_b=*/false, b,
      /*c_mod=*/(short)0, c, /*reuse_a=*/false, /*reuse_b=*/false);
}

// fp32 -> bf16 bits, round-to-nearest-even
__device__ __forceinline__ bfu f2bf(float f) {
  unsigned int u = __float_as_uint(f);
  u += 0x7FFFu + ((u >> 16) & 1u);
  return (bfu)(u >> 16);
}

// ---------------------------------------------------------------------------
// Async global -> LDS copy (GLOBAL_LOAD_ASYNC_TO_LDS_B128, ASYNCcnt-tracked).
// Signature discovered from clang diagnostic:
//   (v4i __device__* gsrc, v4i AS3* lds, imm int offset, imm int cpol)
// AS pointers built via integer round-trips: generic-LDS low 32 bits == LDS
// byte offset; global generic address == AS1 address.
// ---------------------------------------------------------------------------
#if __has_builtin(__builtin_amdgcn_global_load_async_to_lds_b128)
#define HAS_ASYNC_LDS 1
#else
#define HAS_ASYNC_LDS 0
#endif

__device__ __forceinline__ void async_ld_b128(void* lds_dst, const void* gsrc) {
#if HAS_ASYNC_LDS
  __builtin_amdgcn_global_load_async_to_lds_b128(
      (__attribute__((address_space(1))) v4i*)(uintptr_t)gsrc,
      (__attribute__((address_space(3))) v4i*)(unsigned int)(uintptr_t)lds_dst,
      /*offset=*/0, /*cpol=*/0);
#else
  *(uint4*)lds_dst = *(const uint4*)gsrc;
#endif
}

__device__ __forceinline__ void async_wait() {
#if HAS_ASYNC_LDS
#if __has_builtin(__builtin_amdgcn_s_wait_asynccnt)
  __builtin_amdgcn_s_wait_asynccnt(0);
#else
  asm volatile("s_wait_asynccnt 0x0" ::: "memory");
#endif
#endif
}

// ---------------------------------------------------------------------------
// Stage 0: fp32 -> bf16 conversion (vectorized: float4 in, 4x bf16 out)
// ---------------------------------------------------------------------------
__global__ __launch_bounds__(256)
void cvt_f32_to_bf16(const float* __restrict__ in, bfu* __restrict__ out, int n4) {
  int i = blockIdx.x * blockDim.x + threadIdx.x;
  if (i >= n4) return;
  float4 f = ((const float4*)in)[i];
  ushort4 p;
  p.x = f2bf(f.x); p.y = f2bf(f.y); p.z = f2bf(f.z); p.w = f2bf(f.w);
  ((ushort4*)out)[i] = p;
}

// ---------------------------------------------------------------------------
// Stage 1/3: C[M,N] = A[M,K] * W[N,K]^T  (bf16 in, fp32 accumulate)
//   Block: 256 threads / 8 waves, tile 128x128, each wave owns 32x64.
//   Double-buffered async global->LDS staging of 128x32 A/W tiles.
//   MODE 1: fp32 row-major out (final projection -> d_out)
//   MODE 2: bf16 out, head-split [b,h,s,dk]      (Q, K)
//   MODE 3: bf16 out, head-split+T [b,h,dk,s]    (V transposed)
// ---------------------------------------------------------------------------
template <int MODE>
__global__ __launch_bounds__(256)
void gemm_wmma(const bfu* __restrict__ A, const bfu* __restrict__ W,
               void* __restrict__ Cout, int M, int N, int K) {
  __shared__ alignas(16) bfu As[2][128][32];
  __shared__ alignas(16) bfu Bs[2][128][32];

  const int tid  = threadIdx.x;
  const int wid  = tid >> 5;
  const int lane = tid & 31;
  const int lh   = lane & 15;     // row/col within 16-tile
  const int lx   = lane >> 4;     // half-wave select
  const int wm   = wid >> 1;      // 0..3 : 32-row slab
  const int wn   = wid & 1;       // 0..1 : 64-col slab
  const int m0   = blockIdx.x * 128;
  const int n0   = blockIdx.y * 128;

  v8f acc[2][4];
#pragma unroll
  for (int i = 0; i < 2; i++)
#pragma unroll
    for (int j = 0; j < 4; j++) acc[i][j] = v8f_zero();

  // cooperative stage of one 128x32 K-slab of A and W into LDS buffer `buf`
  auto stage = [&](int buf, int kt) {
    const int k0 = kt * 32;
    for (int c = tid; c < 512; c += 256) {
      const int row = c >> 2, kof = (c & 3) * 8;
      async_ld_b128(&As[buf][row][kof], &A[(size_t)(m0 + row) * K + k0 + kof]);
      async_ld_b128(&Bs[buf][row][kof], &W[(size_t)(n0 + row) * K + k0 + kof]);
    }
  };

  const int nkt = K / 32;
  stage(0, 0);

  for (int kt = 0; kt < nkt; ++kt) {
    const int buf = kt & 1;
    async_wait();           // this wave's async LDS writes for tile kt landed
    __syncthreads();        // everyone's landed; old reads of buf^1 retired
    if (kt + 1 < nkt) stage(buf ^ 1, kt + 1);   // overlap DMA with math

    // A-frag: lane = row M, halfs K = {kb..kb+7, kb+16..kb+23}, kb = lx*8
    FragB af[2];
#pragma unroll
    for (int mi = 0; mi < 2; mi++) {
      const int r = wm * 32 + mi * 16 + lh;
      af[mi].q[0] = *(const uint4*)&As[buf][r][lx * 8];
      af[mi].q[1] = *(const uint4*)&As[buf][r][lx * 8 + 16];
    }
    // B-frag: lane = col N, halfs K = {kb..kb+7, kb+16..kb+23}, kb = lx*8
    FragB bfr[4];
#pragma unroll
    for (int ni = 0; ni < 4; ni++) {
      const int r = wn * 64 + ni * 16 + lh;
      bfr[ni].q[0] = *(const uint4*)&Bs[buf][r][lx * 8];
      bfr[ni].q[1] = *(const uint4*)&Bs[buf][r][lx * 8 + 16];
    }
#pragma unroll
    for (int mi = 0; mi < 2; mi++)
#pragma unroll
      for (int ni = 0; ni < 4; ni++)
        acc[mi][ni] = wmma_bf16(af[mi].v, bfr[ni].v, acc[mi][ni]);
  }

  // C layout: VGPR e -> M = lx*8 + e (within tile), N = lh
#pragma unroll
  for (int mi = 0; mi < 2; mi++) {
#pragma unroll
    for (int ni = 0; ni < 4; ni++) {
#pragma unroll
      for (int e = 0; e < 8; e++) {
        const int m = m0 + wm * 32 + mi * 16 + lx * 8 + e;
        const int n = n0 + wn * 64 + ni * 16 + lh;
        const float v = acc[mi][ni][e];
        if constexpr (MODE == 1) {
          ((float*)Cout)[(size_t)m * N + n] = v;
        } else {
          const int b = m / SEQ, s = m % SEQ;
          const int h = n / DK,  d = n % DK;
          size_t idx;
          if constexpr (MODE == 2)
            idx = (((size_t)b * NUM_HEADS + h) * SEQ + s) * DK + d;
          else
            idx = (((size_t)b * NUM_HEADS + h) * DK + d) * SEQ + s;
          ((bfu*)Cout)[idx] = f2bf(v);
        }
      }
    }
  }
}

// ---------------------------------------------------------------------------
// Stage 2: fused flash attention (non-causal).
//   Grid: (B*H, SEQ/128). Block: 256 thr / 8 waves; wave = one 16-query tile.
//   KV tiles of 64 double-buffered in LDS via async loads; online softmax;
//   P->LDS->A-frag; P*V WMMA.
//   Q,K: [b,h,s,dk] ; V: [b,h,dk,s] ; out: [b,s,h*64+dk] bf16
// ---------------------------------------------------------------------------
__global__ __launch_bounds__(256)
void flash_attn(const bfu* __restrict__ Q, const bfu* __restrict__ Kt,
                const bfu* __restrict__ Vt, bfu* __restrict__ O) {
  __shared__ alignas(16) bfu Ks[2][64][64];     // [buf][kv][dk]
  __shared__ alignas(16) bfu Vs[2][64][64];     // [buf][dk][kv]
  __shared__ alignas(16) bfu Ps[8][16][64];     // per-wave P tile [q][kv]

  const int tid  = threadIdx.x;
  const int wid  = tid >> 5;
  const int lane = tid & 31;
  const int lh   = lane & 15, lx = lane >> 4;
  const int bh   = blockIdx.x;
  const int b    = bh / NUM_HEADS, h = bh % NUM_HEADS;
  const int q0   = blockIdx.y * 128 + wid * 16;

  const bfu* Qp = Q  + (size_t)bh * SEQ * DK;
  const bfu* Kp = Kt + (size_t)bh * SEQ * DK;
  const bfu* Vp = Vt + (size_t)bh * DK * SEQ;

  // Q A-frags straight from global (row q = lane, dk halfs {lx*8..+7, lx*8+16..+23})
  FragB qf[2];
  {
    const bfu* qr = Qp + (size_t)(q0 + lh) * DK;
    qf[0].q[0] = *(const uint4*)&qr[lx * 8];
    qf[0].q[1] = *(const uint4*)&qr[lx * 8 + 16];
    qf[1].q[0] = *(const uint4*)&qr[32 + lx * 8];
    qf[1].q[1] = *(const uint4*)&qr[32 + lx * 8 + 16];
  }

  float mrow[8], lrow[8];
#pragma unroll
  for (int e = 0; e < 8; e++) { mrow[e] = -1e30f; lrow[e] = 0.0f; }
  v8f oacc[4];
#pragma unroll
  for (int t = 0; t < 4; t++) oacc[t] = v8f_zero();

  // stage K tile [64 kv][64 dk] and V^T tile [64 dk][64 kv]
  auto stage = [&](int buf, int it) {
    const int kv0 = it * 64;
    for (int c = tid; c < 512; c += 256) {
      const int r = c >> 3, kof = (c & 7) * 8;
      async_ld_b128(&Ks[buf][r][kof], &Kp[(size_t)(kv0 + r) * DK + kof]);
      async_ld_b128(&Vs[buf][r][kof], &Vp[(size_t)r * SEQ + kv0 + kof]);
    }
  };

  const int nit = SEQ / 64;
  stage(0, 0);

  for (int it = 0; it < nit; ++it) {
    const int buf = it & 1;
    async_wait();
    __syncthreads();
    if (it + 1 < nit) stage(buf ^ 1, it + 1);

    // S = Q * K^T : B-frag lane = key index, dk halfs {lx*8..+7, lx*8+16..+23}
    v8f sacc[4];
#pragma unroll
    for (int nt = 0; nt < 4; nt++) sacc[nt] = v8f_zero();
#pragma unroll
    for (int nt = 0; nt < 4; nt++) {
      FragB kf0, kf1;
      const int key = nt * 16 + lh;
      kf0.q[0] = *(const uint4*)&Ks[buf][key][lx * 8];
      kf0.q[1] = *(const uint4*)&Ks[buf][key][lx * 8 + 16];
      kf1.q[0] = *(const uint4*)&Ks[buf][key][32 + lx * 8];
      kf1.q[1] = *(const uint4*)&Ks[buf][key][32 + lx * 8 + 16];
      sacc[nt] = wmma_bf16(qf[0].v, kf0.v, sacc[nt]);
      sacc[nt] = wmma_bf16(qf[1].v, kf1.v, sacc[nt]);
    }

    // online softmax; row m = lx*8 + e lives in one 16-lane half-wave
#pragma unroll
    for (int e = 0; e < 8; e++) {
      float sv[4];
      float mx = -1e30f;
#pragma unroll
      for (int nt = 0; nt < 4; nt++) {
        sv[nt] = sacc[nt][e] * 0.125f;   // 1/sqrt(64)
        mx = fmaxf(mx, sv[nt]);
      }
#pragma unroll
      for (int off = 1; off < 16; off <<= 1)
        mx = fmaxf(mx, __shfl_xor(mx, off, 32));
      const float mnew  = fmaxf(mrow[e], mx);
      const float alpha = __expf(mrow[e] - mnew);
      const int   prow  = lx * 8 + e;
      float rs = 0.0f;
#pragma unroll
      for (int nt = 0; nt < 4; nt++) {
        const float p = __expf(sv[nt] - mnew);
        rs += p;
        Ps[wid][prow][nt * 16 + lh] = f2bf(p);
      }
#pragma unroll
      for (int off = 1; off < 16; off <<= 1)
        rs += __shfl_xor(rs, off, 32);
      lrow[e] = lrow[e] * alpha + rs;
      mrow[e] = mnew;
#pragma unroll
      for (int t = 0; t < 4; t++) oacc[t][e] = oacc[t][e] * alpha;
    }

    // O += P * V : A-frag from Ps, B-frag = V^T row kv {lx*8..+7, lx*8+16..+23}
    FragB pf[2];
#pragma unroll
    for (int j = 0; j < 2; j++) {
      pf[j].q[0] = *(const uint4*)&Ps[wid][lh][j * 32 + lx * 8];
      pf[j].q[1] = *(const uint4*)&Ps[wid][lh][j * 32 + lx * 8 + 16];
    }
#pragma unroll
    for (int t = 0; t < 4; t++) {
#pragma unroll
      for (int j = 0; j < 2; j++) {
        FragB vf;
        const int d = t * 16 + lh;
        vf.q[0] = *(const uint4*)&Vs[buf][d][j * 32 + lx * 8];
        vf.q[1] = *(const uint4*)&Vs[buf][d][j * 32 + lx * 8 + 16];
        oacc[t] = wmma_bf16(pf[j].v, vf.v, oacc[t]);
      }
    }
  }

  // normalize and store merged heads [b, s, h*64+dk] as bf16
#pragma unroll
  for (int t = 0; t < 4; t++) {
#pragma unroll
    for (int e = 0; e < 8; e++) {
      const int q = q0 + lx * 8 + e;
      const int d = t * 16 + lh;
      O[((size_t)b * SEQ + q) * D_MODEL + h * DK + d] =
          f2bf(oacc[t][e] / lrow[e]);
    }
  }
}

// ---------------------------------------------------------------------------
extern "C" void kernel_launch(void* const* d_in, const int* in_sizes, int n_in,
                              void* d_out, int out_size, void* d_ws, size_t ws_size,
                              hipStream_t stream) {
  const float* x  = (const float*)d_in[0];
  const float* Wq = (const float*)d_in[1];
  const float* Wk = (const float*)d_in[2];
  const float* Wv = (const float*)d_in[3];
  const float* Wo = (const float*)d_in[4];

  const size_t M  = (size_t)BATCH * SEQ;        // 8192
  const size_t XE = M * D_MODEL;                // activation elements
  const size_t WE = (size_t)D_MODEL * D_MODEL;  // weight elements

  char* ws = (char*)d_ws;
  bfu* xh  = (bfu*)ws; ws += XE * sizeof(bfu);
  bfu* wqh = (bfu*)ws; ws += WE * sizeof(bfu);
  bfu* wkh = (bfu*)ws; ws += WE * sizeof(bfu);
  bfu* wvh = (bfu*)ws; ws += WE * sizeof(bfu);
  bfu* woh = (bfu*)ws; ws += WE * sizeof(bfu);
  bfu* Qb  = (bfu*)ws; ws += XE * sizeof(bfu);
  bfu* Kb  = (bfu*)ws; ws += XE * sizeof(bfu);
  bfu* Vtb = (bfu*)ws; ws += XE * sizeof(bfu);
  bfu* Ab  = (bfu*)ws; ws += XE * sizeof(bfu);

  // Stage 0: fp32 -> bf16
  cvt_f32_to_bf16<<<(int)(XE / 4 / 256), 256, 0, stream>>>(x,  xh,  (int)(XE / 4));
  cvt_f32_to_bf16<<<(int)(WE / 4 / 256), 256, 0, stream>>>(Wq, wqh, (int)(WE / 4));
  cvt_f32_to_bf16<<<(int)(WE / 4 / 256), 256, 0, stream>>>(Wk, wkh, (int)(WE / 4));
  cvt_f32_to_bf16<<<(int)(WE / 4 / 256), 256, 0, stream>>>(Wv, wvh, (int)(WE / 4));
  cvt_f32_to_bf16<<<(int)(WE / 4 / 256), 256, 0, stream>>>(Wo, woh, (int)(WE / 4));

  // Stage 1: Q/K/V projections (V stored transposed per head)
  dim3 g((unsigned)(M / 128), (unsigned)(D_MODEL / 128));
  gemm_wmma<2><<<g, 256, 0, stream>>>(xh, wqh, Qb,  (int)M, D_MODEL, D_MODEL);
  gemm_wmma<2><<<g, 256, 0, stream>>>(xh, wkh, Kb,  (int)M, D_MODEL, D_MODEL);
  gemm_wmma<3><<<g, 256, 0, stream>>>(xh, wvh, Vtb, (int)M, D_MODEL, D_MODEL);

  // Stage 2: fused flash attention
  flash_attn<<<dim3(BATCH * NUM_HEADS, SEQ / 128), 256, 0, stream>>>(Qb, Kb, Vtb, Ab);

  // Stage 3: output projection -> fp32 d_out
  gemm_wmma<1><<<g, 256, 0, stream>>>(Ab, woh, d_out, (int)M, D_MODEL, D_MODEL);
}
